// DuQuantLinear_52243982189239
// MI455X (gfx1250) — matrix operand order repaired and checked by reading, not compiled, not run on hardware
//
#include <hip/hip_runtime.h>
#include <hip/hip_bf16.h>
#include <stdint.h>

// ---------------------------------------------------------------------------
// DuQuant linear: perm + block-rot -> percentile act-quant (8b) ->
// weight double block-rot + row quant (4b) -> f16 WMMA GEMM -> out-rot + bias
// GEMM: double-buffered LDS; A via global_load_async_to_lds (ASYNCcnt),
//       B via Tensor Data Mover (TENSORcnt); branch-free hot loop.
// ---------------------------------------------------------------------------

typedef __attribute__((ext_vector_type(16))) _Float16 v16h;
typedef __attribute__((ext_vector_type(8)))  float    v8f;
typedef __attribute__((ext_vector_type(4)))  unsigned v4u;
typedef __attribute__((ext_vector_type(8)))  int      v8i_;
typedef __attribute__((ext_vector_type(4)))  int      v4i_;

#define B_    4
#define S_    2048
#define M_    (B_ * S_)      // 8192 rows
#define INF_  4096
#define OUTF_ 4096
#define BLK_  16
#define NBI_  (INF_ / BLK_)
#define NBO_  (OUTF_ / BLK_)
#define QMAXA 127.0f
#define QMAXW 7.0f
#define EPSQ  1e-6f

union frag16 { v16h h; uint4 u[2]; };

// ---- async global -> LDS copy (16B per lane), ASYNCcnt path ---------------
__device__ __forceinline__ void async_copy16(const void* gbase, uint32_t goff,
                                             uint32_t ldsoff) {
  asm volatile("global_load_async_to_lds_b128 %0, %1, %2"
               :: "v"(ldsoff), "v"(goff),
                  "s"((uint64_t)(uintptr_t)gbase)
               : "memory");
}
#define WAIT_ASYNC(n) asm volatile("s_wait_asynccnt " #n ::: "memory")

// ---- Tensor Data Mover 2D tile load (one instruction, wave-level) ---------
#if __has_builtin(__builtin_amdgcn_tensor_load_to_lds)
#define HAVE_TDM 1
#else
#define HAVE_TDM 0
#endif

__device__ __forceinline__ void tdm_load_2d_f16(uint64_t gaddr, uint32_t ldsoff,
                                                uint32_t tensor_d0, uint32_t tensor_d1,
                                                uint32_t tile_d0, uint32_t tile_d1,
                                                uint32_t stride0) {
#if HAVE_TDM
  // D# group 0: count=1 | lds_addr | global_addr[56:0] | type=2
  v4u g0;
  g0[0] = 1u;
  g0[1] = ldsoff;
  g0[2] = (uint32_t)gaddr;
  g0[3] = ((uint32_t)(gaddr >> 32) & 0x01FFFFFFu) | (2u << 30);
  // D# group 1: wg_mask=0, data_size=1 (2B), dims/strides per ISA bit layout
  v8i_ g1;
  g1[0] = (int)(1u << 16);                                   // data_size = 2B
  g1[1] = (int)(tensor_d0 << 16);                            // dim0[15:0] (abar=0)
  g1[2] = (int)((tensor_d0 >> 16) | (tensor_d1 << 16));      // dim0[31:16]|dim1[15:0]
  g1[3] = (int)((tensor_d1 >> 16) | (tile_d0 << 16));        // dim1[31:16]|tile0
  g1[4] = (int)(tile_d1 & 0xFFFFu);                          // tile1 (tile2=0)
  g1[5] = (int)stride0;                                      // dim0_stride[31:0]
  g1[6] = 0;
  g1[7] = 0;
  v4i_ z4 = (v4i_)0;
  v8i_ z8 = (v8i_)0;
  __builtin_amdgcn_tensor_load_to_lds(g0, g1, z4, z4, z8, 0);
#else
  (void)gaddr; (void)ldsoff; (void)tensor_d0; (void)tensor_d1;
  (void)tile_d0; (void)tile_d1; (void)stride0;
#endif
}

// ---------------------------------------------------------------------------
// 0) zero per-row weight absmax
__global__ void k_init(unsigned* wmax) {
  int i = blockIdx.x * 256 + threadIdx.x;
  if (i < OUTF_) wmax[i] = 0u;
}

// ---------------------------------------------------------------------------
// 1) fused weight transform per 16x16 tile + per-row absmax
__global__ __launch_bounds__(256) void k_wt(const float* __restrict__ W,
                                            const int* __restrict__ perm,
                                            const float* __restrict__ Rin,
                                            const float* __restrict__ Rout,
                                            float* __restrict__ Wt,
                                            unsigned* __restrict__ wmax) {
  const int b = blockIdx.x;
  const int c = blockIdx.y;
  const int t = threadIdx.x;
  const int r = t >> 4, q = t & 15;
  __shared__ float Wp[16][16], T[16][16], Ri[16][16], Ro[16][16];
  Ri[r][q] = Rin[(b * 16 + r) * 16 + q];
  Ro[r][q] = Rout[(c * 16 + r) * 16 + q];
  Wp[r][q] = W[(c * 16 + r) * INF_ + perm[b * 16 + q]];
  __syncthreads();
  float acc = 0.f;
#pragma unroll
  for (int m = 0; m < 16; ++m) acc += Wp[r][m] * Ri[m][q];
  T[r][q] = acc;
  __syncthreads();
  float acc2 = 0.f;
#pragma unroll
  for (int k = 0; k < 16; ++k) acc2 += Ro[k][r] * T[k][q];
  Wt[(c * 16 + r) * INF_ + b * 16 + q] = acc2;
  atomicMax(&wmax[c * 16 + r], __float_as_uint(fabsf(acc2)));
}

// ---------------------------------------------------------------------------
// 2) 4-bit weight fake-quant -> f16
__global__ void k_wq(const float* __restrict__ Wt,
                     const unsigned* __restrict__ wmax,
                     _Float16* __restrict__ Wq) {
  int i = blockIdx.x * 256 + threadIdx.x;
  int o = i >> 12;
  float s = fmaxf(__uint_as_float(wmax[o]) / QMAXW, EPSQ);
  float q = fminf(fmaxf(rintf(Wt[i] / s), -QMAXW), QMAXW) * s;
  Wq[i] = (_Float16)q;
}

// ---------------------------------------------------------------------------
// 3) activation transform: xt[r, 16b+j] = sum_m x[r, perm[16b+m]] * Rin[b][m][j]
__global__ __launch_bounds__(256) void k_xt(const float* __restrict__ x,
                                            const int* __restrict__ perm,
                                            const float* __restrict__ Rin,
                                            float* __restrict__ xt) {
  const int row = blockIdx.y;
  const int cb = blockIdx.x;
  __shared__ float xg[256];
  const int t = threadIdx.x;
  xg[t] = x[row * INF_ + perm[cb * 256 + t]];
  __syncthreads();
  const int bl = t >> 4, j = t & 15;
  const int b = cb * 16 + bl;
  float acc = 0.f;
#pragma unroll
  for (int m = 0; m < 16; ++m) acc += xg[bl * 16 + m] * Rin[(b * 16 + m) * 16 + j];
  xt[row * INF_ + cb * 256 + bl * 16 + j] = acc;
}

// ---------------------------------------------------------------------------
// 4) per-channel 99.9-percentile via binary search on float bit patterns
#define KTH_LO 8182
#define QFRAC  0.809f
__global__ __launch_bounds__(256) void k_act_scale(const float* __restrict__ xt,
                                                   float* __restrict__ sa) {
  const int ch = blockIdx.x;
  const int t = threadIdx.x;
  __shared__ unsigned cnt_s, min_s;
  unsigned lo = 0u, hi = 0x7f800000u;
  while (lo < hi) {
    unsigned mid = lo + ((hi - lo) >> 1);
    if (t == 0) cnt_s = 0u;
    __syncthreads();
    unsigned local = 0u;
    for (int r = t; r < M_; r += 256)
      local += (__float_as_uint(fabsf(xt[r * INF_ + ch])) <= mid) ? 1u : 0u;
    atomicAdd(&cnt_s, local);
    __syncthreads();
    unsigned c = cnt_s;
    __syncthreads();
    if (c >= (unsigned)(KTH_LO + 1)) hi = mid; else lo = mid + 1;
  }
  if (t == 0) { cnt_s = 0u; min_s = 0x7f7fffffu; }
  __syncthreads();
  unsigned local = 0u, lmin = 0x7f7fffffu;
  for (int r = t; r < M_; r += 256) {
    unsigned u = __float_as_uint(fabsf(xt[r * INF_ + ch]));
    if (u <= lo) ++local;
    else if (u < lmin) lmin = u;
  }
  atomicAdd(&cnt_s, local);
  atomicMin(&min_s, lmin);
  __syncthreads();
  if (t == 0) {
    float v1 = __uint_as_float(lo);
    float v2 = (cnt_s >= (unsigned)(KTH_LO + 2) || min_s == 0x7f7fffffu)
                   ? v1 : __uint_as_float(min_s);
    float p = v1 * (1.0f - QFRAC) + v2 * QFRAC;
    sa[ch] = fmaxf(p / QMAXA, EPSQ);
  }
}

// ---------------------------------------------------------------------------
// 5) 8-bit activation fake-quant -> f16
__global__ void k_xq(const float* __restrict__ xt,
                     const float* __restrict__ sa,
                     _Float16* __restrict__ xq) {
  int i = blockIdx.x * 256 + threadIdx.x;
  int ch = i & (INF_ - 1);
  float s = sa[ch];
  float q = fminf(fmaxf(rintf(xt[i] / s), -QMAXA), QMAXA) * s;
  xq[i] = (_Float16)q;
}

// ---------------------------------------------------------------------------
// 6) WMMA GEMM: y[M,N] = xq[M,K] * Wq[N,K]^T, f16 in, f32 accumulate.
//    WG tile 128x256, 8 waves (2x4), each wave 64x64 = 4x4 wmma tiles, BK=32.
//    Double-buffered LDS; last K-step peeled so the hot loop is branch-free.
#define BM_ 128
#define BN_ 256
#define BK_ 32
__global__ __launch_bounds__(256) void k_gemm(const _Float16* __restrict__ A,
                                              const _Float16* __restrict__ Bm,
                                              float* __restrict__ C) {
  __shared__ __align__(16) _Float16 As[2][BM_ * BK_];  // [m][k]
  __shared__ __align__(16) _Float16 Bs[2][BN_ * BK_];  // [n][k]
  const int tid = threadIdx.x;
  const int lane = tid & 31;
  const int wave = tid >> 5;
  const int wm = wave >> 2;   // 0..1 -> 64-row strip
  const int wn = wave & 3;    // 0..3 -> 64-col strip
  const int m0 = blockIdx.y * BM_;
  const int n0 = blockIdx.x * BN_;
  const int half = lane >> 4;
  const int l16 = lane & 15;

  uint32_t asOff[2] = {(uint32_t)(uintptr_t)&As[0][0], (uint32_t)(uintptr_t)&As[1][0]};
  uint32_t bsOff[2] = {(uint32_t)(uintptr_t)&Bs[0][0], (uint32_t)(uintptr_t)&Bs[1][0]};

  v8f acc[4][4];
#pragma unroll
  for (int i = 0; i < 4; ++i)
#pragma unroll
    for (int j = 0; j < 4; ++j) acc[i][j] = (v8f){};

  auto stage = [&](int buf, int k0) {
    // A tile 128x32 f16: 512 x 16B lane transfers (2 async issues per wave)
#pragma unroll
    for (int i = 0; i < 2; ++i) {
      int idx = tid + i * 256;
      int row = idx >> 2;
      int c16 = idx & 3;
      async_copy16(A, ((uint32_t)(m0 + row) * INF_ + (uint32_t)k0) * 2u + c16 * 16u,
                   asOff[buf] + (uint32_t)(row * (BK_ * 2) + c16 * 16));
    }
#if HAVE_TDM
    // B tile 256x32 f16: one TDM 2D descriptor, issued by wave 0 only
    if (tid < 32) {
      uint64_t ga = (uint64_t)(uintptr_t)Bm +
                    ((uint64_t)n0 * INF_ + (uint64_t)k0) * 2u;
      tdm_load_2d_f16(ga, bsOff[buf], INF_, OUTF_, BK_, BN_, INF_);
    }
#else
#pragma unroll
    for (int i = 0; i < 4; ++i) {
      int idx = tid + i * 256;
      int row = idx >> 2;
      int c16 = idx & 3;
      async_copy16(Bm, ((uint32_t)(n0 + row) * INF_ + (uint32_t)k0) * 2u + c16 * 16u,
                   bsOff[buf] + (uint32_t)(row * (BK_ * 2) + c16 * 16));
    }
#endif
  };

  auto compute = [&](int buf) {
    frag16 af[4], bf[4];
#pragma unroll
    for (int mt = 0; mt < 4; ++mt) {
      const _Float16* p = &As[buf][0] + (wm * 64 + mt * 16 + l16) * BK_ + half * 8;
      af[mt].u[0] = *(const uint4*)p;
      af[mt].u[1] = *(const uint4*)(p + 16);
    }
#pragma unroll
    for (int nt = 0; nt < 4; ++nt) {
      const _Float16* p = &Bs[buf][0] + (wn * 64 + nt * 16 + l16) * BK_ + half * 8;
      bf[nt].u[0] = *(const uint4*)p;
      bf[nt].u[1] = *(const uint4*)(p + 16);
    }
#pragma unroll
    for (int mt = 0; mt < 4; ++mt)
#pragma unroll
      for (int nt = 0; nt < 4; ++nt)
        acc[mt][nt] = __builtin_amdgcn_wmma_f32_16x16x32_f16(
            false, af[mt].h, false, bf[nt].h, (short)0, acc[mt][nt],
            false, false);
  };

  stage(0, 0);

  // branch-free hot loop: always prefetch next tile, wait on current only
  for (int k0 = 0; k0 < INF_ - BK_; k0 += BK_) {
    const int buf = (k0 / BK_) & 1;
    stage(buf ^ 1, k0 + BK_);
#if HAVE_TDM
    WAIT_ASYNC(2);
    __builtin_amdgcn_s_wait_tensorcnt((short)1);  // no-op for non-issuing waves
#else
    WAIT_ASYNC(6);
#endif
    __syncthreads();
    compute(buf);
    __syncthreads();
  }

  // peeled last K-step
  WAIT_ASYNC(0);
#if HAVE_TDM
  __builtin_amdgcn_s_wait_tensorcnt((short)0);
#endif
  __syncthreads();
  compute(((INF_ / BK_) - 1) & 1);

  // epilogue: VGPR j -> M = j + 8*(lane>=16), N = lane&15
#pragma unroll
  for (int mt = 0; mt < 4; ++mt)
#pragma unroll
    for (int nt = 0; nt < 4; ++nt) {
      int col = n0 + wn * 64 + nt * 16 + l16;
      int rbase = m0 + wm * 64 + mt * 16 + half * 8;
#pragma unroll
      for (int j = 0; j < 8; ++j)
        C[(rbase + j) * OUTF_ + col] = acc[mt][nt][j];
    }
}

// ---------------------------------------------------------------------------
// 7) output block rotation (R_out^T) + bias
__global__ __launch_bounds__(256) void k_yrot(const float* __restrict__ y,
                                              const float* __restrict__ Rout,
                                              const float* __restrict__ bias,
                                              float* __restrict__ out) {
  const int row = blockIdx.y;
  const int cb = blockIdx.x;
  __shared__ float yg[256];
  const int t = threadIdx.x;
  yg[t] = y[row * OUTF_ + cb * 256 + t];
  __syncthreads();
  const int bl = t >> 4, j = t & 15;
  const int c = cb * 16 + bl;
  float acc = bias[c * 16 + j];
#pragma unroll
  for (int k = 0; k < 16; ++k) acc += yg[bl * 16 + k] * Rout[(c * 16 + j) * 16 + k];
  out[row * OUTF_ + cb * 256 + bl * 16 + j] = acc;
}

// ---------------------------------------------------------------------------
extern "C" void kernel_launch(void* const* d_in, const int* in_sizes, int n_in,
                              void* d_out, int out_size, void* d_ws, size_t ws_size,
                              hipStream_t stream) {
  const float* x    = (const float*)d_in[0];
  const float* W    = (const float*)d_in[1];
  const float* bias = (const float*)d_in[2];
  const float* Rin  = (const float*)d_in[3];
  const float* Rout = (const float*)d_in[4];
  const int*   perm = (const int*)d_in[5];

  char* ws = (char*)d_ws;
  const size_t XT_OFF = 0;                                   // M*INF f32 (reused as y_pre)
  const size_t XQ_OFF = XT_OFF + (size_t)M_ * INF_ * 4;
  const size_t WT_OFF = XQ_OFF + (size_t)M_ * INF_ * 2;
  const size_t WQ_OFF = WT_OFF + (size_t)OUTF_ * INF_ * 4;
  const size_t SA_OFF = WQ_OFF + (size_t)OUTF_ * INF_ * 2;
  const size_t WM_OFF = SA_OFF + (size_t)INF_ * 4;

  float*     xt   = (float*)(ws + XT_OFF);
  _Float16*  xq   = (_Float16*)(ws + XQ_OFF);
  float*     Wt   = (float*)(ws + WT_OFF);
  _Float16*  Wq   = (_Float16*)(ws + WQ_OFF);
  float*     sa   = (float*)(ws + SA_OFF);
  unsigned*  wmax = (unsigned*)(ws + WM_OFF);
  float*     ypre = xt;  // reuse
  float*     out  = (float*)d_out;

  k_init<<<(OUTF_ + 255) / 256, 256, 0, stream>>>(wmax);
  k_wt<<<dim3(NBI_, NBO_), 256, 0, stream>>>(W, perm, Rin, Rout, Wt, wmax);
  k_wq<<<((size_t)OUTF_ * INF_) / 256, 256, 0, stream>>>(Wt, wmax, Wq);
  k_xt<<<dim3(INF_ / 256, M_), 256, 0, stream>>>(x, perm, Rin, xt);
  k_act_scale<<<INF_, 256, 0, stream>>>(xt, sa);
  k_xq<<<((size_t)M_ * INF_) / 256, 256, 0, stream>>>(xt, sa, xq);
  k_gemm<<<dim3(OUTF_ / BN_, M_ / BM_), 256, 0, stream>>>(xq, Wq, ypre);
  k_yrot<<<dim3(OUTF_ / 256, M_), 256, 0, stream>>>(ypre, Rout, bias, out);
}
